// Classifier_63136019251669
// MI455X (gfx1250) — matrix operand order repaired and checked by reading, not compiled, not run on hardware
//
#include <hip/hip_runtime.h>
#include <hip/hip_bf16.h>
#include <math.h>

typedef __attribute__((ext_vector_type(16))) _Float16 v16h;
typedef __attribute__((ext_vector_type(8)))  float    v8f;

#define N_NODES   100000
#define N_EDGES   1600000
#define NUM_GRAPHS 64
#define HID        64

// ---------------------------------------------------------------------------
// WMMA GEMM: Y[N,M] = X[N,K] @ W[K,M] (+ optional bias).  One 16x16 tile per
// wave32, K-loop in steps of 32 using v_wmma_f32_16x16x32_f16 (f32 accumulate).
// Requires N%16==0, K%32==0, M%16==0 (true for all call sites here).
// ---------------------------------------------------------------------------
__global__ void gemm_wmma_f16(const float* __restrict__ X, const float* __restrict__ W,
                              const float* __restrict__ bias, float* __restrict__ Y,
                              int Nrows, int K, int M)
{
    const int lane = threadIdx.x & 31;
    const int hl   = lane >> 4;        // lane half: 0 = lanes 0-15, 1 = lanes 16-31
    const int l16  = lane & 15;
    const int wavesPerBlock = blockDim.x >> 5;
    const int waveId = blockIdx.x * wavesPerBlock + (threadIdx.x >> 5);
    const int tilesM = M >> 4;
    const int tilesN = Nrows >> 4;
    if (waveId >= tilesM * tilesN) return;   // wave-uniform guard: EXEC all-1s at WMMA
    const int rowTile = waveId / tilesM;
    const int colTile = waveId % tilesM;
    const int row0 = rowTile << 4;
    const int col0 = colTile << 4;

    v8f c = {};
    for (int k = 0; k < K; k += 32) {
        // A tile 16x32 f16 (ISA 7.12.2): lane half selects K sub-range
        v16h a, b;
        const float* ap = X + (size_t)(row0 + l16) * K + k + hl * 8;
        #pragma unroll
        for (int i = 0; i < 8; ++i) {
            a[i]     = (_Float16)ap[i];        // K = k + hl*8 + i
            a[i + 8] = (_Float16)ap[i + 16];   // K = k + hl*8 + 16 + i
        }
        // B tile 32x16 f16: same K order as A (K = k + hl*8 + j, then k + hl*8 + 16 + j)
        const float* bp = W + (size_t)(k + hl * 8) * M + col0 + l16;
        #pragma unroll
        for (int j = 0; j < 16; ++j) b[j] = (_Float16)bp[(size_t)(j + (j & 8)) * M];
        c = __builtin_amdgcn_wmma_f32_16x16x32_f16(false, a, false, b,
                                                   (short)0, c, false, false);
    }
    const float bv = bias ? bias[col0 + l16] : 0.0f;
    float* yp = Y + (size_t)(row0 + hl * 8) * M + col0 + l16;
    #pragma unroll
    for (int r = 0; r < 8; ++r) yp[(size_t)r * M] = c[r] + bv;   // D row = row0+hl*8+r
}

// ---------------------------------------------------------------------------
// agg[dst] += w_e * t[src]  for 64-dim features; 16 lanes per edge, float4 each
// ---------------------------------------------------------------------------
__global__ void scatter_weighted64(const float* __restrict__ t, const float* __restrict__ ew,
                                   const int* __restrict__ src, const int* __restrict__ dst,
                                   float* __restrict__ agg)
{
    long long gidx = (long long)blockIdx.x * blockDim.x + threadIdx.x;
    int e = (int)(gidx >> 4);
    if (e >= N_EDGES) return;
    int c = ((int)gidx & 15) << 2;
    int s = src[e], d = dst[e];
    float w = ew[e];
    float4 v = *(const float4*)(t + (size_t)s * HID + c);
    float* o = agg + (size_t)d * HID + c;
    atomicAdd(o + 0, w * v.x);
    atomicAdd(o + 1, w * v.y);
    atomicAdd(o + 2, w * v.z);
    atomicAdd(o + 3, w * v.w);
}

__global__ void bias_relu64(float* __restrict__ x, const float* __restrict__ b, long long n)
{
    long long i = (long long)blockIdx.x * blockDim.x + threadIdx.x;
    if (i >= n) return;
    float v = x[i] + b[(int)i & 63];
    x[i] = v > 0.0f ? v : 0.0f;
}

__global__ void fill_f32(float* __restrict__ p, float v, long long n)
{
    long long i = (long long)blockIdx.x * blockDim.x + threadIdx.x;
    if (i < n) p[i] = v;
}

// ---------------------------------------------------------------------------
// el[n,h] = sum_d feat[n,h,d]*attn_l[h,d]; er likewise. One wave per node:
// lane covers 4 dims of the flat 128; lanes 0-15 -> head0, 16-31 -> head1.
// ---------------------------------------------------------------------------
__global__ void gat_scores(const float* __restrict__ feat, const float* __restrict__ al,
                           const float* __restrict__ ar, float* __restrict__ el,
                           float* __restrict__ er)
{
    int node = blockIdx.x * (blockDim.x >> 5) + (threadIdx.x >> 5);
    int lane = threadIdx.x & 31;
    if (node >= N_NODES) return;
    float4 f = *(const float4*)(feat + (size_t)node * 128 + lane * 4);
    float4 a = *(const float4*)(al + lane * 4);
    float4 b = *(const float4*)(ar + lane * 4);
    float sl = f.x * a.x + f.y * a.y + f.z * a.z + f.w * a.w;
    float sr = f.x * b.x + f.y * b.y + f.z * b.z + f.w * b.w;
    #pragma unroll
    for (int m = 8; m >= 1; m >>= 1) {     // reduce within 16-lane halves
        sl += __shfl_xor(sl, m, 32);
        sr += __shfl_xor(sr, m, 32);
    }
    if ((lane & 15) == 0) {
        int h = lane >> 4;
        el[(size_t)node * 2 + h] = sl;
        er[(size_t)node * 2 + h] = sr;
    }
}

// Pass 1: e = leaky_relu(el[src]+er[dst]); m[dst] = max(m[dst], e)
__global__ void gat_edge_max(const float* __restrict__ el, const float* __restrict__ er,
                             const int* __restrict__ src, const int* __restrict__ dst,
                             float* __restrict__ ebuf, float* __restrict__ m)
{
    long long gidx = (long long)blockIdx.x * blockDim.x + threadIdx.x;
    int e = (int)(gidx >> 1);
    if (e >= N_EDGES) return;
    int h = (int)gidx & 1;
    int s = src[e], d = dst[e];
    float v = el[(size_t)s * 2 + h] + er[(size_t)d * 2 + h];
    v = v > 0.0f ? v : 0.2f * v;
    ebuf[(size_t)e * 2 + h] = v;
    int* mi = (int*)(m + (size_t)d * 2 + h);        // float max via bit tricks
    if (v >= 0.0f) atomicMax(mi, __float_as_int(v));
    else           atomicMin((unsigned int*)mi, __float_as_uint(v));
}

// Pass 2: a = exp(e - m[dst]) (in place); z[dst] += a
__global__ void gat_edge_exp(float* __restrict__ ebuf, const float* __restrict__ m,
                             const int* __restrict__ dst, float* __restrict__ z)
{
    long long gidx = (long long)blockIdx.x * blockDim.x + threadIdx.x;
    int e = (int)(gidx >> 1);
    if (e >= N_EDGES) return;
    int h = (int)gidx & 1;
    int d = dst[e];
    float a = expf(ebuf[(size_t)e * 2 + h] - m[(size_t)d * 2 + h]);
    ebuf[(size_t)e * 2 + h] = a;
    atomicAdd(&z[(size_t)d * 2 + h], a);
}

// Pass 3: acc[dst,d] += sum_h (a/z * w_e) * feat[src, h*64+d]; 16 lanes/edge
__global__ void gat_apply(const float* __restrict__ feat, const float* __restrict__ abuf,
                          const float* __restrict__ z, const float* __restrict__ ew,
                          const int* __restrict__ src, const int* __restrict__ dst,
                          float* __restrict__ acc)
{
    long long gidx = (long long)blockIdx.x * blockDim.x + threadIdx.x;
    int e = (int)(gidx >> 4);
    if (e >= N_EDGES) return;
    int l = (int)gidx & 15;
    int s = src[e], d = dst[e];
    float w  = ew[e];
    float c0 = abuf[(size_t)e * 2 + 0] / z[(size_t)d * 2 + 0] * w;
    float c1 = abuf[(size_t)e * 2 + 1] / z[(size_t)d * 2 + 1] * w;
    int dim = l << 2;
    float4 f0 = *(const float4*)(feat + (size_t)s * 128 + dim);
    float4 f1 = *(const float4*)(feat + (size_t)s * 128 + 64 + dim);
    float* o = acc + (size_t)d * HID + dim;
    atomicAdd(o + 0, c0 * f0.x + c1 * f1.x);
    atomicAdd(o + 1, c0 * f0.y + c1 * f1.y);
    atomicAdd(o + 2, c0 * f0.z + c1 * f1.z);
    atomicAdd(o + 3, c0 * f0.w + c1 * f1.w);
}

// hn = relu((acc + bg_h0 + bg_h1)/2); hg[graph] += hn  (sum pooling)
__global__ void pool_nodes(const float* __restrict__ acc, const float* __restrict__ bg,
                           const int* __restrict__ gids, float* __restrict__ hg)
{
    long long gidx = (long long)blockIdx.x * blockDim.x + threadIdx.x;
    if (gidx >= (long long)N_NODES * HID) return;
    int n  = (int)(gidx >> 6);
    int dm = (int)gidx & 63;
    float v = (acc[gidx] + bg[dm] + bg[64 + dm]) * 0.5f;
    v = v > 0.0f ? v : 0.0f;
    atomicAdd(&hg[(size_t)gids[n] * HID + dm], v);
}

// Per-column training-mode BatchNorm (biased var) + ReLU, batch of 64 rows
__global__ void bn_relu(const float* __restrict__ x, const float* __restrict__ gamma,
                        const float* __restrict__ beta, float* __restrict__ y, int cols)
{
    int j = blockIdx.x * blockDim.x + threadIdx.x;
    if (j >= cols) return;
    float mu = 0.0f;
    for (int r = 0; r < NUM_GRAPHS; ++r) mu += x[(size_t)r * cols + j];
    mu *= (1.0f / NUM_GRAPHS);
    float var = 0.0f;
    for (int r = 0; r < NUM_GRAPHS; ++r) {
        float d = x[(size_t)r * cols + j] - mu;
        var += d * d;
    }
    var *= (1.0f / NUM_GRAPHS);
    float inv = rsqrtf(var + 1e-5f);
    float g = gamma[j], b = beta[j];
    for (int r = 0; r < NUM_GRAPHS; ++r) {
        float v = g * (x[(size_t)r * cols + j] - mu) * inv + b;
        y[(size_t)r * cols + j] = v > 0.0f ? v : 0.0f;
    }
}

// logits = z2 @ Wc3 + bc3; softmax per row
__global__ void head_softmax(const float* __restrict__ z2, const float* __restrict__ Wc3,
                             const float* __restrict__ bc3, float* __restrict__ out)
{
    int r = threadIdx.x;
    if (r >= NUM_GRAPHS) return;
    float l0 = bc3[0], l1 = bc3[1];
    for (int j = 0; j < 128; ++j) {
        float v = z2[(size_t)r * 128 + j];
        l0 += v * Wc3[j * 2 + 0];
        l1 += v * Wc3[j * 2 + 1];
    }
    float mx = fmaxf(l0, l1);
    float e0 = expf(l0 - mx), e1 = expf(l1 - mx);
    float s = e0 + e1;
    out[r * 2 + 0] = e0 / s;
    out[r * 2 + 1] = e1 / s;
}

// ---------------------------------------------------------------------------
extern "C" void kernel_launch(void* const* d_in, const int* in_sizes, int n_in,
                              void* d_out, int out_size, void* d_ws, size_t ws_size,
                              hipStream_t stream)
{
    const float* h      = (const float*)d_in[0];
    const float* ew     = (const float*)d_in[1];
    const int*   src    = (const int*)d_in[2];
    const int*   dst    = (const int*)d_in[3];
    const int*   gids   = (const int*)d_in[4];
    const float* W1     = (const float*)d_in[5];
    const float* b1     = (const float*)d_in[6];
    const float* W2     = (const float*)d_in[7];
    const float* b2     = (const float*)d_in[8];
    const float* Wg     = (const float*)d_in[9];
    const float* attn_l = (const float*)d_in[10];
    const float* attn_r = (const float*)d_in[11];
    const float* bg     = (const float*)d_in[12];
    const float* Wc1    = (const float*)d_in[13];
    const float* bc1    = (const float*)d_in[14];
    const float* g1     = (const float*)d_in[15];
    const float* be1    = (const float*)d_in[16];
    const float* Wc2    = (const float*)d_in[17];
    const float* bc2    = (const float*)d_in[18];
    const float* g2     = (const float*)d_in[19];
    const float* be2    = (const float*)d_in[20];
    const float* Wc3    = (const float*)d_in[21];
    const float* bc3    = (const float*)d_in[22];
    float* out = (float*)d_out;

    char* ws = (char*)d_ws;
    size_t off = 0;
    auto alloc = [&](size_t bytes) -> float* {
        float* p = (float*)(ws + off);
        off += (bytes + 255) & ~(size_t)255;
        return p;
    };
    const size_t nodeHidB = (size_t)N_NODES * HID * sizeof(float);
    float* bufA = alloc(nodeHidB);                       // t1, then t2, then acc
    float* bufB = alloc(nodeHidB);                       // agg1/h1, then agg2/h2
    float* feat = alloc((size_t)N_NODES * 128 * sizeof(float));
    float* ebuf = alloc((size_t)N_EDGES * 2 * sizeof(float));
    float* el   = alloc((size_t)N_NODES * 2 * sizeof(float));
    float* er   = alloc((size_t)N_NODES * 2 * sizeof(float));
    float* mbuf = alloc((size_t)N_NODES * 2 * sizeof(float));
    float* zbuf = alloc((size_t)N_NODES * 2 * sizeof(float));
    float* hg   = alloc((size_t)NUM_GRAPHS * HID * sizeof(float));
    float* y1   = alloc((size_t)NUM_GRAPHS * 256 * sizeof(float));
    float* z1   = alloc((size_t)NUM_GRAPHS * 256 * sizeof(float));
    float* y2   = alloc((size_t)NUM_GRAPHS * 128 * sizeof(float));
    float* z2   = alloc((size_t)NUM_GRAPHS * 128 * sizeof(float));

    auto gemm = [&](const float* X, const float* W, const float* bias, float* Y,
                    int N, int K, int M) {
        int waves  = (N >> 4) * (M >> 4);
        int blocks = (waves + 7) / 8;                    // 8 waves (256 thr) per block
        gemm_wmma_f16<<<blocks, 256, 0, stream>>>(X, W, bias, Y, N, K, M);
    };

    const int eB16 = (int)(((long long)N_EDGES * 16 + 255) / 256);  // 16 lanes/edge
    const int eB2  = (int)(((long long)N_EDGES * 2 + 255) / 256);   // 2 thr/edge
    const int nB64 = (int)(((long long)N_NODES * HID + 255) / 256);

    // ---- GraphConv 1 (linear hoisted before aggregation) ----
    gemm(h, W1, nullptr, bufA, N_NODES, 128, HID);                       // t1 = h@W1
    hipMemsetAsync(bufB, 0, nodeHidB, stream);
    scatter_weighted64<<<eB16, 256, 0, stream>>>(bufA, ew, src, dst, bufB);
    bias_relu64<<<nB64, 256, 0, stream>>>(bufB, b1, (long long)N_NODES * HID);  // h1

    // ---- GraphConv 2 ----
    gemm(bufB, W2, nullptr, bufA, N_NODES, HID, HID);                    // t2 = h1@W2
    hipMemsetAsync(bufB, 0, nodeHidB, stream);      // h1 dead; reuse for agg2
    scatter_weighted64<<<eB16, 256, 0, stream>>>(bufA, ew, src, dst, bufB);
    bias_relu64<<<nB64, 256, 0, stream>>>(bufB, b2, (long long)N_NODES * HID);  // h2

    // ---- GATConv ----
    gemm(bufB, Wg, nullptr, feat, N_NODES, HID, 128);                    // feat = h2@Wg
    gat_scores<<<(N_NODES + 7) / 8, 256, 0, stream>>>(feat, attn_l, attn_r, el, er);
    fill_f32<<<(2 * N_NODES + 255) / 256, 256, 0, stream>>>(mbuf, -INFINITY,
                                                            (long long)2 * N_NODES);
    hipMemsetAsync(zbuf, 0, (size_t)N_NODES * 2 * sizeof(float), stream);
    gat_edge_max<<<eB2, 256, 0, stream>>>(el, er, src, dst, ebuf, mbuf);
    gat_edge_exp<<<eB2, 256, 0, stream>>>(ebuf, mbuf, dst, zbuf);
    hipMemsetAsync(bufA, 0, nodeHidB, stream);      // t2 dead; reuse for acc
    gat_apply<<<eB16, 256, 0, stream>>>(feat, ebuf, zbuf, ew, src, dst, bufA);

    // ---- graph sum pooling ----
    hipMemsetAsync(hg, 0, (size_t)NUM_GRAPHS * HID * sizeof(float), stream);
    pool_nodes<<<nB64, 256, 0, stream>>>(bufA, bg, gids, hg);

    // ---- classifier MLP (WMMA gemms, 64 rows) ----
    gemm(hg, Wc1, bc1, y1, NUM_GRAPHS, HID, 256);
    bn_relu<<<1, 256, 0, stream>>>(y1, g1, be1, z1, 256);
    gemm(z1, Wc2, bc2, y2, NUM_GRAPHS, 256, 128);
    bn_relu<<<1, 128, 0, stream>>>(y2, g2, be2, z2, 128);
    head_softmax<<<1, 64, 0, stream>>>(z2, Wc3, bc3, out);
}
